// SelfAttention_9637906612422
// MI455X (gfx1250) — matrix operand order repaired and checked by reading, not compiled, not run on hardware
//
#include <hip/hip_runtime.h>

typedef __attribute__((ext_vector_type(16))) __bf16 v16bf;
typedef __attribute__((ext_vector_type(8)))  __bf16 v8bf;
typedef __attribute__((ext_vector_type(8)))  float  v8f;

#define S_LEN 2048
#define DKDIM 128
#define KT_STRIDE 40   // padded row stride (ushorts) for transposed K tile
#define QK_SCALE 0.08838834764831845f  // 1/sqrt(128)

static __device__ __forceinline__ v16bf bfcat(v8bf lo, v8bf hi) {
    return __builtin_shufflevector(lo, hi, 0, 1, 2, 3, 4, 5, 6, 7,
                                           8, 9, 10, 11, 12, 13, 14, 15);
}
static __device__ __forceinline__ unsigned short bf16b(float f) {
    return __builtin_bit_cast(unsigned short, (__bf16)f);
}

// One flash-attention path: Out (+)= outScale * softmax(Q K^T / sqrt(dk)) V
// Block = 256 threads (8 waves). Each wave owns a 32-row query tile (two
// 16-row WMMA row-tiles that share every K/V B-fragment); the block
// cooperatively streams 32-key K/V tiles through LDS as bf16.
// K is staged TRANSPOSED (d-major); each K B-fragment element is a
// separate 16-bit LDS read, one kT row (d) per element. The softmax denominator is
// kept as per-lane partial sums and reduced once in the epilogue.
__global__ __launch_bounds__(256, 1)
void diffattn_flash_path(const float* __restrict__ Q,
                         const float* __restrict__ K,
                         const float* __restrict__ V,
                         float* __restrict__ Out,
                         float outScale, int accumulate)
{
    __shared__ __align__(16) unsigned short kT[DKDIM * KT_STRIDE];   // 10 KB, kT[d][key]
    __shared__ __align__(16) unsigned short vTile[32 * DKDIM];       // 8 KB, row-major
    __shared__ __align__(16) unsigned short pStage[8 * 2 * 16 * 32]; // 16 KB, per-wave 2x(16x32)

    const int tid  = threadIdx.x;
    const int lane = tid & 31;
    const int wave = tid >> 5;
    const int bh   = blockIdx.y;                  // 0 .. B*H-1
    const int qbase = blockIdx.x * 256 + wave * 32;

    const float* Qb = Q + (size_t)bh * S_LEN * DKDIM;
    const float* Kb = K + (size_t)bh * S_LEN * DKDIM;
    const float* Vb = V + (size_t)bh * S_LEN * DKDIM;
    float*       Ob = Out + (size_t)bh * S_LEN * DKDIM;

    const int m     = lane & 15;   // row (A) / column (B,C) index within tile
    const int hi    = lane >> 4;   // half-wave select
    const int kbase = hi * 8;      // A-layout K base for this half

    // ---- Q tile -> 2 row-tiles x 4 bf16 A-fragments, 1/sqrt(dk) folded in ----
    v16bf qf[2][4];
#pragma unroll
    for (int u = 0; u < 2; ++u) {
#pragma unroll
        for (int c = 0; c < 4; ++c) {
#pragma unroll
            for (int e = 0; e < 16; ++e) {
                const int d = c * 32 + kbase + ((e < 8) ? e : (e + 8));
                qf[u][c][e] =
                    (__bf16)(Qb[(size_t)(qbase + u * 16 + m) * DKDIM + d] * QK_SCALE);
            }
        }
    }

    v8f acc[2][8];
#pragma unroll
    for (int u = 0; u < 2; ++u)
#pragma unroll
        for (int c = 0; c < 8; ++c) acc[u][c] = v8f{};
    float mrun[2][8], lpart[2][8];
#pragma unroll
    for (int u = 0; u < 2; ++u)
#pragma unroll
        for (int r = 0; r < 8; ++r) { mrun[u][r] = -1e30f; lpart[u][r] = 0.0f; }

    unsigned short* pMine = &pStage[wave * (2 * 16 * 32)];

    for (int kt = 0; kt < S_LEN; kt += 32) {
        // ================= cooperative staging =================
        {
            // --- K: load 4 keys x 4 d per thread, store transposed ---
            const int dg = tid & 31;    // d block: d = dg*4 (coalesced across lanes)
            const int kg = tid >> 5;    // key block: key = kg*4
            float4 kv[4];
#pragma unroll
            for (int i = 0; i < 4; ++i)
                kv[i] = ((const float4*)(Kb + (size_t)(kt + kg * 4 + i) * DKDIM))[dg];
#pragma unroll
            for (int j = 0; j < 4; ++j) {
                const unsigned int lo =
                    (unsigned int)bf16b(((const float*)&kv[0])[j]) |
                    ((unsigned int)bf16b(((const float*)&kv[1])[j]) << 16);
                const unsigned int hi2 =
                    (unsigned int)bf16b(((const float*)&kv[2])[j]) |
                    ((unsigned int)bf16b(((const float*)&kv[3])[j]) << 16);
                uint2 val; val.x = lo; val.y = hi2;
                *(uint2*)&kT[(dg * 4 + j) * KT_STRIDE + kg * 4] = val;  // ds_store_b64
            }

            // --- V: row-major, 16 contiguous floats per thread ---
            const int row = tid >> 3;          // 0..31
            const int col = (tid & 7) * 16;    // 0..112
            float4 vv[4];
#pragma unroll
            for (int i = 0; i < 4; ++i)
                vv[i] = ((const float4*)(Vb + (size_t)(kt + row) * DKDIM + col))[i];
            unsigned short vb[16];
#pragma unroll
            for (int i = 0; i < 4; ++i) {
                const float* vf = (const float*)&vv[i];
#pragma unroll
                for (int j = 0; j < 4; ++j)
                    vb[i * 4 + j] = bf16b(vf[j]);
            }
#pragma unroll
            for (int i = 0; i < 2; ++i)
                ((uint4*)&vTile[row * DKDIM + col])[i] = ((const uint4*)vb)[i];  // ds_store_b128

            // --- prefetch next tile into cache hierarchy ---
            if (kt + 32 < S_LEN) {
                __builtin_prefetch(Kb + (size_t)(kt + 32 + kg * 4) * DKDIM + dg * 4, 0, 1);
                __builtin_prefetch(Vb + (size_t)(kt + 32 + row) * DKDIM + col, 0, 1);
            }
        }
        __syncthreads();

        // ---- S = Q K^T : each K fragment feeds BOTH row-tiles ----
        v8f s[2][2];   // [row-tile][key-tile]
#pragma unroll
        for (int u = 0; u < 2; ++u)
#pragma unroll
            for (int t = 0; t < 2; ++t) s[u][t] = v8f{};
#pragma unroll
        for (int t = 0; t < 2; ++t) {
#pragma unroll
            for (int c = 0; c < 4; ++c) {
                v16bf kf;
#pragma unroll
                for (int e = 0; e < 16; ++e)
                    kf[e] = __builtin_bit_cast(__bf16, kT[(c * 32 + kbase + ((e < 8) ? e : (e + 8))) * KT_STRIDE + t * 16 + m]);
#pragma unroll
                for (int u = 0; u < 2; ++u)
                    s[u][t] = __builtin_amdgcn_wmma_f32_16x16x32_bf16(
                                false, qf[u][c], false, kf, (short)0, s[u][t], false, false);
            }
        }

        // ---- online softmax: per-step row-max butterfly, deferred sum ----
#pragma unroll
        for (int u = 0; u < 2; ++u) {
#pragma unroll
            for (int r = 0; r < 8; ++r) {
                const float s0 = s[u][0][r];
                const float s1 = s[u][1][r];
                float tm = fmaxf(s0, s1);
#pragma unroll
                for (int off = 1; off < 16; off <<= 1)
                    tm = fmaxf(tm, __shfl_xor(tm, off, 32));
                const float mnew  = fmaxf(mrun[u][r], tm);
                const float alpha = __expf(mrun[u][r] - mnew);
                const float p0 = __expf(s0 - mnew);
                const float p1 = __expf(s1 - mnew);
                lpart[u][r] = lpart[u][r] * alpha + (p0 + p1);
                mrun[u][r]  = mnew;
#pragma unroll
                for (int c = 0; c < 8; ++c) acc[u][c][r] *= alpha;
                // stage P: C layout -> LDS row-major (16x32 bf16 per row-tile)
                const int row = r + 8 * hi;
                pMine[u * 512 + row * 32 + m]      = bf16b(p0);
                pMine[u * 512 + row * 32 + 16 + m] = bf16b(p1);
            }
        }
        asm volatile("s_wait_dscnt 0" ::: "memory");

        // ---- P A-fragments (two contiguous 16B halves each) ----
        v16bf pf[2];
#pragma unroll
        for (int u = 0; u < 2; ++u)
            pf[u] = bfcat(*(const v8bf*)&pMine[u * 512 + m * 32 + kbase],
                          *(const v8bf*)&pMine[u * 512 + m * 32 + kbase + 16]);

        // ---- O += P V : each V fragment feeds BOTH row-tiles ----
#pragma unroll
        for (int c = 0; c < 8; ++c) {
            v16bf vf;
#pragma unroll
            for (int e = 0; e < 16; ++e)
                vf[e] = __builtin_bit_cast(__bf16, vTile[(kbase + ((e < 8) ? e : (e + 8))) * DKDIM + c * 16 + m]);
#pragma unroll
            for (int u = 0; u < 2; ++u)
                acc[u][c] = __builtin_amdgcn_wmma_f32_16x16x32_bf16(
                              false, pf[u], false, vf, (short)0, acc[u][c], false, false);
        }
        __syncthreads();
    }

    // ---- epilogue: one cross-lane reduction of each denominator ----
#pragma unroll
    for (int u = 0; u < 2; ++u) {
#pragma unroll
        for (int r = 0; r < 8; ++r) {
            float rs = lpart[u][r];
#pragma unroll
            for (int off = 1; off < 16; off <<= 1)
                rs += __shfl_xor(rs, off, 32);
            const float inv = 1.0f / rs;
#pragma unroll
            for (int c = 0; c < 8; ++c) {
                const int row = u * 16 + r + 8 * hi;
                const float o = acc[u][c][r] * inv * outScale;
                const size_t idx = (size_t)(qbase + row) * DKDIM + c * 16 + m;
                if (accumulate) Ob[idx] += o;
                else            Ob[idx]  = o;
            }
        }
    }
}

extern "C" void kernel_launch(void* const* d_in, const int* in_sizes, int n_in,
                              void* d_out, int out_size, void* d_ws, size_t ws_size,
                              hipStream_t stream) {
    (void)in_sizes; (void)n_in; (void)out_size; (void)d_ws; (void)ws_size;
    const float* q  = (const float*)d_in[0];
    const float* k  = (const float*)d_in[1];
    const float* v  = (const float*)d_in[2];
    const float* dq = (const float*)d_in[3];
    const float* dk = (const float*)d_in[4];
    float* out = (float*)d_out;

    dim3 grid(S_LEN / 256, 2 * 16);   // 8 query-tiles-of-256 x (B*H)
    dim3 block(256);

    // Pass 1: out = softmax(QK^T)V
    diffattn_flash_path<<<grid, block, 0, stream>>>(q, k, v, out, 1.0f, 0);
    // Pass 2: out -= 0.5 * softmax(dQ dK^T)V
    diffattn_flash_path<<<grid, block, 0, stream>>>(dq, dk, v, out, -0.5f, 1);
}
